// MixedScoreMHA_7928509628562
// MI455X (gfx1250) — matrix operand order repaired and checked by reading, not compiled, not run on hardware
//
#include <hip/hip_runtime.h>
#include <cmath>

typedef __attribute__((ext_vector_type(16))) __bf16 v16bf;
typedef __attribute__((ext_vector_type(8)))  __bf16 v8bf;
typedef __attribute__((ext_vector_type(4)))  __bf16 v4bf;
typedef __attribute__((ext_vector_type(8)))  float  v8f;
typedef __attribute__((ext_vector_type(4)))  float  v4f;

#define WMMA_BF16(a, b, c) \
  __builtin_amdgcn_wmma_f32_16x16x32_bf16(false, (a), false, (b), (short)0, (c), false, false)

#define CAT16(lo, hi) __builtin_shufflevector((lo), (hi), 0,1,2,3,4,5,6,7,8,9,10,11,12,13,14,15)

static constexpr int Bn = 32, Rn = 512, Cn = 512, Dn = 256, Hn = 16, KDn = 16;

// ---------------------------------------------------------------------------
// fp32 -> bf16 elementwise convert (vector of 4 per thread)
// ---------------------------------------------------------------------------
__global__ __launch_bounds__(256) void cvt_bf16_kernel(const float* __restrict__ src,
                                                       __bf16* __restrict__ dst, int n4) {
  const int i = blockIdx.x * 256 + threadIdx.x;
  if (i < n4) {
    v4f x = *(const v4f*)(src + (size_t)i * 4);
    v4bf y;
    #pragma unroll
    for (int e = 0; e < 4; ++e) y[e] = (__bf16)x[e];
    *(v4bf*)(dst + (size_t)i * 4) = y;
  }
}

// ---------------------------------------------------------------------------
// Pack a 256x256 fp32 weight into bf16 B-fragment layout:
//   Wp[((nT*8 + kc)*32 + lane)*16 + e] = W[(kc*32 + 8*khi + (e < 8 ? e : e + 8))*256 + nT*16 + l15]
// so each wave's B fragment is one contiguous 32B load per lane.
// ---------------------------------------------------------------------------
__global__ __launch_bounds__(32) void packw_kernel(const float* __restrict__ W,
                                                   __bf16* __restrict__ Wp) {
  const int lane = threadIdx.x & 31;
  const int l15 = lane & 15, khi = lane >> 4;
  const int nT = blockIdx.x, kc = blockIdx.y;
  __bf16* dst = Wp + (((size_t)nT * 8 + kc) * 32 + lane) * 16;
  const float* src = W + (size_t)(kc * 32 + 8 * khi) * Dn + nT * 16 + l15;
  #pragma unroll
  for (int e = 0; e < 16; ++e) dst[e] = (__bf16)src[(size_t)(e < 8 ? e : e + 8) * Dn];
}

// ---------------------------------------------------------------------------
// 512x512 fp32 transpose per batch (LDS tiled, coalesced both ways)
// ---------------------------------------------------------------------------
__global__ __launch_bounds__(256) void transpose512_kernel(const float* __restrict__ in,
                                                           float* __restrict__ out) {
  __shared__ float tile[32][33];
  const int b = blockIdx.z;
  const int x0 = blockIdx.x * 32, y0 = blockIdx.y * 32;
  const float* src = in + (size_t)b * Rn * Cn;
  float* dst = out + (size_t)b * Rn * Cn;
  for (int j = threadIdx.y; j < 32; j += 8)
    tile[j][threadIdx.x] = src[(size_t)(y0 + j) * Cn + x0 + threadIdx.x];
  __syncthreads();
  for (int j = threadIdx.y; j < 32; j += 8)
    dst[(size_t)(x0 + j) * Rn + y0 + threadIdx.x] = tile[threadIdx.x][j];
}

// ---------------------------------------------------------------------------
// Projection: out = embB (B*N,256)bf16 @ Wp(packed). 16x64 tile per wave.
// transposeOut=0: out[b][h][n][kd]   (Q/K layout)
// transposeOut=1: out[b][h][kd][n]   (V layout, b128 stores)
// ---------------------------------------------------------------------------
__global__ __launch_bounds__(256) void proj_kernel(const __bf16* __restrict__ embB,
                                                   const __bf16* __restrict__ Wp,
                                                   __bf16* __restrict__ out,
                                                   int transposeOut) {
  const int lane = threadIdx.x & 31;
  const int l15 = lane & 15, khi = lane >> 4;
  const int mT = blockIdx.x * 8 + threadIdx.y;       // 16-row tile
  const int nG = blockIdx.y;                         // 4 heads per wave
  const size_t rowIdx = (size_t)mT * 16 + l15;
  v8f acc[4] = {};
  for (int kc = 0; kc < 8; ++kc) {
    const __bf16* pa = embB + rowIdx * Dn + kc * 32 + 8 * khi;
    v8bf a0 = *(const v8bf*)pa;
    v8bf a1 = *(const v8bf*)(pa + 16);
    v16bf a = CAT16(a0, a1);
    #pragma unroll
    for (int j = 0; j < 4; ++j) {
      const int nT = nG * 4 + j;
      v16bf bf = *(const v16bf*)(Wp + (((size_t)nT * 8 + kc) * 32 + lane) * 16);
      acc[j] = WMMA_BF16(a, bf, acc[j]);
    }
  }
  const int b = (mT * 16) >> 9;
  #pragma unroll
  for (int j = 0; j < 4; ++j) {
    const int h = nG * 4 + j;
    if (!transposeOut) {
      #pragma unroll
      for (int v = 0; v < 8; ++v) {
        const int n = (mT * 16 + v + 8 * khi) & 511;
        out[(((size_t)b * Hn + h) * 512 + n) * KDn + l15] = (__bf16)acc[j][v];
      }
    } else {
      v8bf st;
      #pragma unroll
      for (int v = 0; v < 8; ++v) st[v] = (__bf16)acc[j][v];
      const int n0 = ((mT * 16) & 511) + 8 * khi;
      *(v8bf*)(out + (((size_t)b * Hn + h) * KDn + l15) * 512 + n0) = st;
    }
  }
}

// ---------------------------------------------------------------------------
// Fused online-softmax attention pass (no LDS, no barriers).
// Q' = [Qa|Qb] (N,32), K' = [Ka|Kb] (N,32), Vt = V transposed [kd][n],
// bias[attRow][attCol] (pre-transposed for the col pass).
// Computes S^T = K' Q'^T so the C-fragment (lane = attRow) is directly the
// A-fragment element pattern for P @ V.
// ---------------------------------------------------------------------------
__global__ __launch_bounds__(256) void attn_kernel(
    const __bf16* __restrict__ Qa, const __bf16* __restrict__ Qb,
    const __bf16* __restrict__ Ka, const __bf16* __restrict__ Kb,
    const __bf16* __restrict__ Vt, const float* __restrict__ bias,
    const float* __restrict__ Wmat, __bf16* __restrict__ heads) {
  const int lane = threadIdx.x & 31;
  const int l15 = lane & 15, khi = lane >> 4;
  const int rT = blockIdx.x * 8 + threadIdx.y;
  const int h = blockIdx.y, b = blockIdx.z;
  const size_t bh = (size_t)b * Hn + h;
  const float norm = 0.25f;                           // 1/sqrt(KD)
  const float nw = norm * Wmat[h];

  // Q' B-fragment (loaded once): lane = r, elements k = 16*khi + e
  const v16bf qf = *(const v16bf*)((khi ? Qb : Qa) + (bh * 512 + (size_t)rT * 16 + l15) * KDn);

  const float* biasRow = bias + ((size_t)b * 512 + (size_t)rT * 16 + l15) * 512;
  const __bf16* vBase = Vt + (bh * KDn + l15) * 512;

  v8f o = {};
  float mOld = -INFINITY, lSum = 0.f;

  for (int it = 0; it < 512 / 32; ++it) {
    if (it + 1 < 512 / 32) {
      __builtin_prefetch((const void*)(Ka + (bh * 512 + (size_t)(it + 1) * 32 + l15) * KDn), 0, 1);
      __builtin_prefetch((const void*)(vBase + (it + 1) * 32), 0, 1);
      __builtin_prefetch((const void*)(biasRow + (it + 1) * 32), 0, 1);
    }
    float cv[16];   // compat, cv[s*8+v] <-> c = it*32 + s*16 + v + 8*khi
    #pragma unroll
    for (int s = 0; s < 2; ++s) {
      const int c0 = it * 32 + s * 16;
      // K' A-fragment: lane = c_local, elements k = 16*khi + e
      v16bf kf = *(const v16bf*)((khi ? Kb : Ka) + (bh * 512 + (size_t)c0 + l15) * KDn);
      v8f st = {};
      st = WMMA_BF16(kf, qf, st);                 // S^T: lane = r, vgpr v -> c
      v4f m0 = *(const v4f*)(biasRow + c0 + 8 * khi);
      v4f m1 = *(const v4f*)(biasRow + c0 + 8 * khi + 4);
      #pragma unroll
      for (int v = 0; v < 8; ++v) {
        const float mv = (v < 4) ? m0[v] : m1[v - 4];
        cv[s * 8 + v] = fmaf(mv, nw, norm * st[v]);
      }
    }
    // online softmax over c (per attention row r = l15)
    float mt = cv[0];
    #pragma unroll
    for (int e = 1; e < 16; ++e) mt = fmaxf(mt, cv[e]);
    mt = fmaxf(mt, __shfl_xor(mt, 16, 32));       // merge khi halves
    const float mNew = fmaxf(mOld, mt);
    const float alpha = __expf(mOld - mNew);
    v16bf pf; float ls = 0.f;
    #pragma unroll
    for (int e = 0; e < 16; ++e) {                // element e == A-frag slot e
      const float p = __expf(cv[e] - mNew);
      ls += p; pf[e] = (__bf16)p;
    }
    ls += __shfl_xor(ls, 16, 32);
    lSum = lSum * alpha + ls;
    mOld = mNew;
    // V B-fragment: lane = kd, elements c = it*32 + 8*khi + e / 16 + 8*khi + (e-8)
    v16bf vf = CAT16(*(const v8bf*)(vBase + (size_t)it * 32 + 8 * khi), *(const v8bf*)(vBase + (size_t)it * 32 + 8 * khi + 16));
    #pragma unroll
    for (int v = 0; v < 8; ++v) o[v] *= __shfl(alpha, v + 8 * khi, 32);
    o = WMMA_BF16(pf, vf, o);
  }
  #pragma unroll
  for (int v = 0; v < 8; ++v) {
    const float li = __shfl(lSum, v + 8 * khi, 32);
    const int rg = rT * 16 + v + 8 * khi;
    heads[((size_t)b * 512 + rg) * (Hn * KDn) + h * KDn + l15] = (__bf16)(o[v] / li);
  }
}

// ---------------------------------------------------------------------------
// Output projection: out_f32 = heads_bf16 (B*N,256) @ Wp(packed). 16x64/wave.
// ---------------------------------------------------------------------------
__global__ __launch_bounds__(256) void outproj_kernel(const __bf16* __restrict__ heads,
                                                      const __bf16* __restrict__ Wp,
                                                      float* __restrict__ out) {
  const int lane = threadIdx.x & 31;
  const int l15 = lane & 15, khi = lane >> 4;
  const int mT = blockIdx.x * 8 + threadIdx.y;
  const int nG = blockIdx.y;
  const size_t rowIdx = (size_t)mT * 16 + l15;
  v8f acc[4] = {};
  for (int kc = 0; kc < 8; ++kc) {
    const __bf16* pa = heads + rowIdx * (Hn * KDn) + kc * 32 + 8 * khi;
    v16bf a = CAT16(*(const v8bf*)pa, *(const v8bf*)(pa + 16));
    #pragma unroll
    for (int j = 0; j < 4; ++j) {
      const int nT = nG * 4 + j;
      v16bf bf = *(const v16bf*)(Wp + (((size_t)nT * 8 + kc) * 32 + lane) * 16);
      acc[j] = WMMA_BF16(a, bf, acc[j]);
    }
  }
  #pragma unroll
  for (int j = 0; j < 4; ++j) {
    const int nT = nG * 4 + j;
    #pragma unroll
    for (int v = 0; v < 8; ++v) {
      const int m = mT * 16 + v + 8 * khi;
      out[(size_t)m * Dn + nT * 16 + l15] = acc[j][v];
    }
  }
}

// ---------------------------------------------------------------------------
extern "C" void kernel_launch(void* const* d_in, const int* in_sizes, int n_in,
                              void* d_out, int out_size, void* d_ws, size_t ws_size,
                              hipStream_t stream) {
  const float* row_emb   = (const float*)d_in[0];
  const float* col_emb   = (const float*)d_in[1];
  const float* matrix    = (const float*)d_in[2];
  const float* W_mat     = (const float*)d_in[9];
  float* out = (float*)d_out;

  // ---- workspace layout (bf16 elements unless noted) ----
  __bf16* ws = (__bf16*)d_ws;
  const size_t NE = (size_t)Bn * 512 * Dn;            // 4M elems
  const size_t WPE = (size_t)Hn * 8 * 32 * 16;        // 65536 elems per packed weight
  __bf16* rowEmbB = ws;
  __bf16* colEmbB = ws + NE;
  __bf16* wpack   = ws + 2 * NE;                      // 8 packed weights
  __bf16* qkv     = ws + 2 * NE + 8 * WPE;            // 6 tensors of NE
  __bf16* rowQ = qkv;          __bf16* rowK = qkv + NE;     __bf16* rowVt = qkv + 2 * NE;
  __bf16* colQ = qkv + 3 * NE; __bf16* colK = qkv + 4 * NE; __bf16* colVt = qkv + 5 * NE;
  __bf16* headsR = qkv + 6 * NE;
  __bf16* headsC = qkv + 7 * NE;
  float* matT = (float*)(qkv + 8 * NE);               // 32 MB fp32

  // ---- 0) one-time repacks ----
  cvt_bf16_kernel<<<dim3((unsigned)(NE / 4 / 256)), dim3(256), 0, stream>>>(row_emb, rowEmbB, (int)(NE / 4));
  cvt_bf16_kernel<<<dim3((unsigned)(NE / 4 / 256)), dim3(256), 0, stream>>>(col_emb, colEmbB, (int)(NE / 4));
  for (int w = 0; w < 8; ++w) {
    static const int srcIdx[8] = {3, 4, 5, 6, 7, 8, 10, 11}; // Wq/k/v_row, Wq/k/v_col, Wout_row, Wout_col
    packw_kernel<<<dim3(16, 8), dim3(32), 0, stream>>>((const float*)d_in[srcIdx[w]], wpack + (size_t)w * WPE);
  }
  transpose512_kernel<<<dim3(16, 16, Bn), dim3(32, 8), 0, stream>>>(matrix, matT);

  // ---- 1) projections ----
  dim3 blk(32, 8, 1);
  dim3 gProj(Bn * 512 / 16 / 8, 4, 1);
  proj_kernel<<<gProj, blk, 0, stream>>>(rowEmbB, wpack + 0 * WPE, rowQ, 0);
  proj_kernel<<<gProj, blk, 0, stream>>>(rowEmbB, wpack + 1 * WPE, rowK, 0);
  proj_kernel<<<gProj, blk, 0, stream>>>(rowEmbB, wpack + 2 * WPE, rowVt, 1);
  proj_kernel<<<gProj, blk, 0, stream>>>(colEmbB, wpack + 3 * WPE, colQ, 0);
  proj_kernel<<<gProj, blk, 0, stream>>>(colEmbB, wpack + 4 * WPE, colK, 0);
  proj_kernel<<<gProj, blk, 0, stream>>>(colEmbB, wpack + 5 * WPE, colVt, 1);

  // ---- 2) fused attention passes ----
  //  row pass:  Q'=[row_q|row_k], K'=[col_k|col_q], V=col_v, bias = matrix
  //  col pass:  Q'=[col_k|col_q], K'=[row_q|row_k], V=row_v, bias = matrix^T
  dim3 gAttn(Rn / 16 / 8, Hn, Bn);
  attn_kernel<<<gAttn, blk, 0, stream>>>(rowQ, rowK, colK, colQ, colVt, matrix, W_mat, headsR);
  attn_kernel<<<gAttn, blk, 0, stream>>>(colK, colQ, rowQ, rowK, rowVt, matT, W_mat, headsC);

  // ---- 3) output projections -> fp32 d_out ----
  dim3 gOut(Bn * 512 / 16 / 8, 4, 1);
  outproj_kernel<<<gOut, blk, 0, stream>>>(headsR, wpack + 6 * WPE, out);
  outproj_kernel<<<gOut, blk, 0, stream>>>(headsC, wpack + 7 * WPE, out + (size_t)Bn * 512 * Dn);
}
